// AFNO2D_20572893348324
// MI455X (gfx1250) — matrix operand order repaired and checked by reading, not compiled, not run on hardware
//
#include <hip/hip_runtime.h>
#include <hip/hip_bf16.h>

// ---- CDNA5 WMMA types -------------------------------------------------------
typedef _Float16 h4  __attribute__((ext_vector_type(4)));
typedef _Float16 h8  __attribute__((ext_vector_type(8)));
typedef _Float16 h16 __attribute__((ext_vector_type(16)));
typedef float    f8  __attribute__((ext_vector_type(8)));

// Generic tiled GEMM: OUT[M x N] = A[M x K](row major) * B[K x N]
// M tiled 128/WG (8 waves x 16 rows), N tiled 64/WG, K in 64-chunks with
// DOUBLE-BUFFERED LDS: async loads for chunk i+1 are issued right after the
// top-of-loop barrier, compute runs on chunk i, ASYNCcnt waited next iter.
//  AF16: A is f16 row-major -> GLOBAL_LOAD_ASYNC_TO_LDS_B128 (16B/lane).
//        else A is f32      -> float4 loads, cvt, packed h4 LDS stores.
//  BT  : B is f16 *pre-transposed* (N x K row-major, ldb = K) -> async copy.
//        else B is f32 K x N row-major -> float4 loads + transposed b16 stores.
// MODE: 0 = none, 1 = relu(acc+bias) -> f16, 2 = softshrink(acc+bias),
//       3 = acc*scale + residual.
template <bool AF16, bool BT, bool OF16, int MODE>
__global__ __launch_bounds__(256) void gemm_wmma(
    const void* __restrict__ Ab, const void* __restrict__ Bb,
    void* __restrict__ Cb,
    const float* __restrict__ bias, int biasStride,
    const float* __restrict__ resid,
    int lda, int ldb, int ldc, int K,
    long long aBatch, long long bBatch, long long cBatch,
    float scale)
{
    __shared__ __attribute__((aligned(16))) _Float16 sA[2][128 * 72];
    __shared__ __attribute__((aligned(16))) _Float16 sB[2][64 * 72];

    const int t    = threadIdx.x;
    const int lane = t & 31;
    const int wave = t >> 5;
    const int mBase = blockIdx.y * 128;
    const int nBase = blockIdx.x * 64;
    const long long z    = blockIdx.z;
    const long long aOff = z * aBatch;
    const long long bOff = z * bBatch;
    const long long cOff = z * cBatch;

    const float*    Af = (const float*)Ab;
    const _Float16* Ah = (const _Float16*)Ab;
    const float*    Bf = (const float*)Bb;
    const _Float16* Bh = (const _Float16*)Bb;

    auto stageA = [&](int p, int kc) {
        if constexpr (AF16) {
            // straight copy: 128 rows x 8 x 16B chunks = 1024, 4 per thread
#pragma unroll
            for (int i = 0; i < 4; ++i) {
                int c16 = t + i * 256;
                int r = c16 >> 3, c = (c16 & 7) * 8;
                const _Float16* gp = &Ah[aOff + (long long)(mBase + r) * lda + kc + c];
                unsigned lp = (unsigned)(unsigned long long)&sA[p][r * 72 + c];
                asm volatile("global_load_async_to_lds_b128 %0, %1, off"
                             :: "v"(lp), "v"(gp) : "memory");
            }
        } else {
#pragma unroll
            for (int i = 0; i < 8; ++i) {
                int q = t + i * 256;               // 16 float4 per row, 128 rows
                int r = q >> 4, c = (q & 15) * 4;
                float4 v = *(const float4*)&Af[aOff + (long long)(mBase + r) * lda + kc + c];
                h4 hv = { (_Float16)v.x, (_Float16)v.y, (_Float16)v.z, (_Float16)v.w };
                *(h4*)&sA[p][r * 72 + c] = hv;
            }
        }
    };
    auto stageB = [&](int p, int kc) {
        if constexpr (BT) {
            // straight copy: 64 n-rows x 8 x 16B chunks = 512, 2 per thread
#pragma unroll
            for (int i = 0; i < 2; ++i) {
                int c16 = t + i * 256;
                int n = c16 >> 3, c = (c16 & 7) * 8;
                const _Float16* gp = &Bh[bOff + (long long)(nBase + n) * ldb + kc + c];
                unsigned lp = (unsigned)(unsigned long long)&sB[p][n * 72 + c];
                asm volatile("global_load_async_to_lds_b128 %0, %1, off"
                             :: "v"(lp), "v"(gp) : "memory");
            }
        } else {
            // f32 K x N: float4 along n, transpose into LDS with b16 stores
#pragma unroll
            for (int i = 0; i < 4; ++i) {
                int q = t + i * 256;               // 64 k-rows x 16 float4 groups
                int k = q >> 4, n0 = (q & 15) * 4;
                float4 v = *(const float4*)&Bf[bOff + (long long)(kc + k) * ldb + nBase + n0];
                sB[p][(n0 + 0) * 72 + k] = (_Float16)v.x;
                sB[p][(n0 + 1) * 72 + k] = (_Float16)v.y;
                sB[p][(n0 + 2) * 72 + k] = (_Float16)v.z;
                sB[p][(n0 + 3) * 72 + k] = (_Float16)v.w;
            }
        }
    };

    f8 acc[4] = {};

    // Prologue: fill buffer 0.
    stageA(0, 0);
    stageB(0, 0);

    const int nk = K >> 6;
    for (int ic = 0; ic < nk; ++ic) {
        // Current buffer's async fills were issued last iteration (or prologue).
        if constexpr (AF16 || BT)
            asm volatile("s_wait_asynccnt 0x0" ::: "memory");
        __syncthreads();

        // Kick off the next chunk into the other buffer (latency hidden by WMMAs).
        if (ic + 1 < nk) {
            stageA((ic + 1) & 1, (ic + 1) * 64);
            stageB((ic + 1) & 1, (ic + 1) * 64);
        }

        const int p = ic & 1;
        // Preload all fragments so the ds_load_b128s pipeline ahead of the WMMAs.
        h16 afr[2];
        h16 bfr[2][4];
#pragma unroll
        for (int ks = 0; ks < 2; ++ks) {
            int k0 = ks * 32;
            int am = wave * 16 + (lane & 15);
            int ak = k0 + ((lane & 16) ? 8 : 0);
            h8 alo = *(const h8*)&sA[p][am * 72 + ak];
            h8 ahi = *(const h8*)&sA[p][am * 72 + ak + 16];
            afr[ks] = __builtin_shufflevector(alo, ahi,
                0, 1, 2, 3, 4, 5, 6, 7, 8, 9, 10, 11, 12, 13, 14, 15);
#pragma unroll
            for (int nt = 0; nt < 4; ++nt) {
                int bn = nt * 16 + (lane & 15);
                int bk = k0 + ((lane & 16) ? 8 : 0);
                h8 blo = *(const h8*)&sB[p][bn * 72 + bk];
                h8 bhi = *(const h8*)&sB[p][bn * 72 + bk + 16];
                bfr[ks][nt] = __builtin_shufflevector(blo, bhi,
                    0, 1, 2, 3, 4, 5, 6, 7, 8, 9, 10, 11, 12, 13, 14, 15);
            }
        }
#pragma unroll
        for (int ks = 0; ks < 2; ++ks)
#pragma unroll
            for (int nt = 0; nt < 4; ++nt)
                acc[nt] = __builtin_amdgcn_wmma_f32_16x16x32_f16(
                    false, afr[ks], false, bfr[ks][nt], (short)0, acc[nt],
                    false, false);
        // No trailing barrier: next iteration's top barrier protects buffer reuse.
    }

    // Epilogue. C/D layout: VGPR v -> row m0 + v (+8 for lanes 16-31), col = lane&15.
    const int mw = mBase + wave * 16 + ((lane & 16) ? 8 : 0);
#pragma unroll
    for (int nt = 0; nt < 4; ++nt) {
        int col = nBase + nt * 16 + (lane & 15);
        float bv = (MODE == 1 || MODE == 2) ? bias[(int)z * biasStride + col] : 0.f;
#pragma unroll
        for (int v = 0; v < 8; ++v) {
            int row = mw + v;
            long long ci = cOff + (long long)row * ldc + col;
            float val = acc[nt][v];
            if (MODE == 1) {
                val += bv;
                val = fmaxf(val, 0.f);
            } else if (MODE == 2) {
                val += bv;
                const float lam = 0.01f;
                val = (val > lam) ? (val - lam) : ((val < -lam) ? (val + lam) : 0.f);
            } else if (MODE == 3) {
                val = val * scale + resid[ci];
            }
            if (OF16) ((_Float16*)Cb)[ci] = (_Float16)val;
            else      ((float*)Cb)[ci] = val;
        }
    }
}

// Precompute the 128x128 Hartley (cas) matrix in f16.
__global__ void k_cas(_Float16* __restrict__ cas) {
    int idx = blockIdx.x * 256 + threadIdx.x;
    if (idx < 128 * 128) {
        int i = idx >> 7, j = idx & 127;
        int p = (i * j) & 127;                 // exact phase mod N keeps precision
        float ang = (float)p * 0.04908738521234052f;  // 2*pi/128
        cas[idx] = (_Float16)(cosf(ang) + sinf(ang));
    }
}

// Effective weights, transposed: d[blk][n][k] = (f16)(s[blk][k][n] + s2[blk][k][n]).
__global__ void k_addcvt_t(const float* __restrict__ s, _Float16* __restrict__ d,
                           int K, int N, int off2) {
    int i = blockIdx.x * 256 + threadIdx.x;   // over 8*K*N
    int per = K * N;
    if (i < 8 * per) {
        int blk = i / per, r = i - blk * per;
        int k = r / N, n = r - k * N;
        d[blk * per + n * K + k] = (_Float16)(s[i] + s[i + off2]);
    }
}

extern "C" void kernel_launch(void* const* d_in, const int* in_sizes, int n_in,
                              void* d_out, int out_size, void* d_ws, size_t ws_size,
                              hipStream_t stream) {
    (void)in_sizes; (void)n_in; (void)out_size; (void)ws_size;
    const float* x  = (const float*)d_in[0];  // (4,16384,1024) = (4,128,128,1024)
    const float* w1 = (const float*)d_in[1];  // (2,8,128,256)
    const float* b1 = (const float*)d_in[2];  // (2,8,256) -> use first half
    const float* w2 = (const float*)d_in[3];  // (2,8,256,128)
    const float* b2 = (const float*)d_in[4];  // (2,8,128) -> use first half
    float* out = (float*)d_out;

    char* ws = (char*)d_ws;
    const long long TEN = 4LL * 16384 * 1024;          // 67,108,864 elements
    float*    buf1 = (float*)ws;                        // 256 MB
    float*    buf2 = (float*)(ws + TEN * 4);            // 256 MB
    _Float16* o1   = (_Float16*)ws;                     // f16, reuses buf1 region
    _Float16* cas  = (_Float16*)(ws + 2 * TEN * 4);     // 32 KB
    _Float16* W1t  = cas + 128 * 128;                   // 8 x 256 x 128 halves (N x K)
    _Float16* W2t  = W1t + 8 * 256 * 128;               // 8 x 128 x 256 halves (N x K)

    // ---- prep ----
    k_cas<<<64, 256, 0, stream>>>(cas);
    k_addcvt_t<<<(262144 + 255) / 256, 256, 0, stream>>>(w1, W1t, 128, 256, 262144);
    k_addcvt_t<<<(262144 + 255) / 256, 256, 0, stream>>>(w2, W2t, 256, 128, 262144);

    // ---- 1) DHT along H: buf1[b,h',w,c] = sum_h cas[h',h] x[b,h,w,c] ----
    gemm_wmma<true, false, false, 0><<<dim3(2048, 1, 4), 256, 0, stream>>>(
        cas, x, buf1, nullptr, 0, nullptr,
        128, 131072, 131072, 128, 0LL, 16777216LL, 16777216LL, 1.f);

    // ---- 2) DHT along W: buf2[b,h,w',c] = sum_w cas[w',w] buf1[b,h,w,c] ----
    gemm_wmma<true, false, false, 0><<<dim3(16, 1, 512), 256, 0, stream>>>(
        cas, buf1, buf2, nullptr, 0, nullptr,
        128, 1024, 1024, 128, 0LL, 131072LL, 131072LL, 1.f);

    // ---- 3) block MLP layer 1: o1[n,p,:] = relu(freq @ W1 + b1[0,n]) -> f16 ----
    // per block n: M=65536(points), K=128, N=256; B = W1t (256 x 128, ldb=K=128)
    gemm_wmma<false, true, true, 1><<<dim3(4, 512, 8), 256, 0, stream>>>(
        buf2, W1t, o1, b1, 256, nullptr,
        1024, 128, 256, 128, 128LL, 32768LL, 16777216LL, 1.f);

    // ---- 4) layer 2 + soft-shrink -> buf2[p, n*128+i] ----
    // per block n: M=65536, K=256, N=128; B = W2t (128 x 256, ldb=K=256)
    gemm_wmma<true, true, false, 2><<<dim3(2, 512, 8), 256, 0, stream>>>(
        o1, W2t, buf2, b2, 128, nullptr,
        256, 256, 1024, 256, 16777216LL, 32768LL, 128LL, 1.f);

    // ---- 5) inverse DHT along W: buf1 = cas @ buf2 (per b,h) ----
    gemm_wmma<true, false, false, 0><<<dim3(16, 1, 512), 256, 0, stream>>>(
        cas, buf2, buf1, nullptr, 0, nullptr,
        128, 1024, 1024, 128, 0LL, 131072LL, 131072LL, 1.f);

    // ---- 6) inverse DHT along H, scale 1/(H*W), add residual x -> out ----
    gemm_wmma<true, false, false, 3><<<dim3(2048, 1, 4), 256, 0, stream>>>(
        cas, buf1, out, nullptr, 0, x,
        128, 131072, 131072, 128, 0LL, 16777216LL, 16777216LL, 1.0f / 16384.0f);
}
